// MambaTSNetwork_52656299049432
// MI455X (gfx1250) — matrix operand order repaired and checked by reading, not compiled, not run on hardware
//
#include <hip/hip_runtime.h>
#include <hip/hip_bf16.h>
#include <math.h>

// ---------------------------------------------------------------------------
// MambaTS forward for MI455X (gfx1250, wave32, WMMA).
// Dense GEMMs: v_wmma_f32_16x16x32_bf16 (f32 accumulate).
// Selective scan: async global->LDS DMA prefetch of B/C rows.
// ---------------------------------------------------------------------------

typedef __attribute__((ext_vector_type(16))) __bf16 v16bf;
typedef __attribute__((ext_vector_type(8)))  float  v8f;
typedef unsigned int u32;
typedef __attribute__((ext_vector_type(8)))  u32    v8u;

#define L_CTX     512
#define K_VAR     16
#define H_PRED    96
#define D_MODEL   512
#define N_STATE   16
#define D_FF      2048
#define D_INNER   1024
#define DT_RANK   32
#define P_PATCH   63
#define TOK_PER_B 1008              // K_VAR * P_PATCH
#define BATCH     16
#define T_TOT     16128             // BATCH * TOK_PER_B
#define EPSF      1e-5f

#define ACT_NONE     0
#define ACT_SILU     1
#define ACT_GELU     2
#define ACT_SOFTPLUS 3

__device__ __forceinline__ float apply_act(float x, int act) {
  if (act == ACT_SILU)     return x / (1.f + __expf(-x));
  if (act == ACT_GELU)     return 0.5f * x * (1.f + erff(x * 0.70710678118654752f));
  if (act == ACT_SOFTPLUS) return (x > 20.f) ? x : log1pf(__expf(x));
  return x;
}

__device__ __forceinline__ u32 pkbf(float lo, float hi) {
  unsigned short l = __builtin_bit_cast(unsigned short, (__bf16)lo);
  unsigned short h = __builtin_bit_cast(unsigned short, (__bf16)hi);
  return ((u32)h << 16) | (u32)l;
}

__device__ __forceinline__ float block_sum(float v, float* sbuf) {
  int tid = threadIdx.x;
  sbuf[tid] = v;
  __syncthreads();
  for (int s = 64; s > 0; s >>= 1) {
    if (tid < s) sbuf[tid] += sbuf[tid + s];
    __syncthreads();
  }
  float r = sbuf[0];
  __syncthreads();
  return r;
}

// ---------------------------------------------------------------------------
// bf16-WMMA GEMM:  C[M,N] = epilogue( A[M,lda(>=K)] @ W[N,K]^T )
// 64x64 block tile, 128 threads (4 waves). Wave w owns M-rows [w*16, w*16+16),
// all four 16-wide N-tiles. K staged 64 at a time (2 WMMA k-steps, 8 WMMAs).
// LDS as packed bf16 pairs (u32), row stride 33 u32 (odd bank stride ->
// conflict-free fragment reads). Per k-step, the A fragment and all four B
// fragments are fetched into distinct registers first (one dscnt wait), then
// the four WMMAs issue back-to-back.
// ---------------------------------------------------------------------------
#define TM   64
#define TN   64
#define TKS  64
#define USTR 33

template <bool GUARD>
__global__ __launch_bounds__(128)
void gemm_bf16_wmma(const float* __restrict__ A, const float* __restrict__ W,
                    const float* __restrict__ bias, const float* __restrict__ res,
                    float* __restrict__ C,
                    int M, int N, int K, int lda, int act)
{
  __shared__ u32 Abuf[TM * USTR];
  __shared__ u32 Wbuf[TN * USTR];

  const int tid  = threadIdx.x;
  const int lane = tid & 31;
  const int wave = tid >> 5;
  const int m0   = blockIdx.y * TM;
  const int n0   = blockIdx.x * TN;
  const int mloc = lane & 15;
  const int half = lane >> 4;

  // staging: thread owns row sr (0..63), 32-col chunk sc (0 or 32)
  const int sr = tid >> 1;
  const int sc = (tid & 1) * 32;
  const float* aptr = A + (size_t)(m0 + sr) * lda + sc;
  const float* wptr = W + (size_t)(n0 + sr) * K + sc;

  // hoisted fragment base pointers
  const u32* apB = &Abuf[(wave * 16 + mloc) * USTR];
  const u32* wpB[4];
  #pragma unroll
  for (int j = 0; j < 4; ++j) wpB[j] = &Wbuf[(j * 16 + mloc) * USTR + (half << 3)];

  v8f zero = {0.f,0.f,0.f,0.f,0.f,0.f,0.f,0.f};
  v8f acc[4] = {zero, zero, zero, zero};

  float4 ar[8], wr[8];

  auto load_tiles = [&](int k0) {
    #pragma unroll
    for (int q = 0; q < 8; ++q) {
      if (!GUARD) {
        ar[q] = *(const float4*)(aptr + k0 + 4 * q);
        wr[q] = *(const float4*)(wptr + k0 + 4 * q);
      } else {
        float4 z4; z4.x = z4.y = z4.z = z4.w = 0.f;
        bool cok = (k0 + sc + 4 * q) < K;
        float4 av = z4, wv = z4;
        if (cok && (m0 + sr) < M) av = *(const float4*)(aptr + k0 + 4 * q);
        if (cok && (n0 + sr) < N) wv = *(const float4*)(wptr + k0 + 4 * q);
        ar[q] = av; wr[q] = wv;
      }
    }
  };
  auto store_tiles = [&]() {
    u32* ab = &Abuf[sr * USTR + (sc >> 1)];
    u32* wb = &Wbuf[sr * USTR + (sc >> 1)];
    #pragma unroll
    for (int q = 0; q < 8; ++q) {
      ab[2 * q]     = pkbf(ar[q].x, ar[q].y);
      ab[2 * q + 1] = pkbf(ar[q].z, ar[q].w);
      wb[2 * q]     = pkbf(wr[q].x, wr[q].y);
      wb[2 * q + 1] = pkbf(wr[q].z, wr[q].w);
    }
  };

  load_tiles(0);
  for (int k0 = 0; k0 < K; k0 += TKS) {
    __syncthreads();                    // previous-stage readers done
    store_tiles();
    __syncthreads();
    if (k0 + TKS < K) load_tiles(k0 + TKS);         // register prefetch (+1)
    if (!GUARD && k0 + 2 * TKS < K) {               // L2 prefetch (+2)
      __builtin_prefetch(aptr + k0 + 2 * TKS, 0, 3);
      __builtin_prefetch(wptr + k0 + 2 * TKS, 0, 3);
    }

    #pragma unroll
    for (int ks = 0; ks < 2; ++ks) {
      // A fragment (16-bit A 16x32): pair idx = half*8 + v (contiguous k, matches B)
      v8u au;
      #pragma unroll
      for (int v = 0; v < 8; ++v)
        au[v] = apB[ks * 16 + (half << 3) + v];

      // all four B fragments up-front (distinct regs -> one dscnt wait,
      // then four back-to-back WMMAs)
      v8u bu[4];
      #pragma unroll
      for (int j = 0; j < 4; ++j) {
        #pragma unroll
        for (int v = 0; v < 8; ++v) bu[j][v] = wpB[j][ks * 16 + v];
      }

      v16bf afrag = __builtin_bit_cast(v16bf, au);
      #pragma unroll
      for (int j = 0; j < 4; ++j) {
        v16bf bfrag = __builtin_bit_cast(v16bf, bu[j]);
        acc[j] = __builtin_amdgcn_wmma_f32_16x16x32_bf16(
            false, afrag, false, bfrag, (short)0, acc[j], false, false);
      }
    }
  }

  // epilogue: C/D layout — VGPR r, lane l: row = r + 8*(l>>4), col = l&15
  #pragma unroll
  for (int j = 0; j < 4; ++j) {
    int col = n0 + j * 16 + mloc;
    if (GUARD && col >= N) continue;
    float bv = bias ? bias[col] : 0.f;
    #pragma unroll
    for (int r = 0; r < 8; ++r) {
      int row = m0 + wave * 16 + r + (half << 3);
      if (GUARD && row >= M) continue;
      float v = apply_act(acc[j][r] + bv, act);
      if (res) v += res[(size_t)row * N + col];
      C[(size_t)row * N + col] = v;
    }
  }
}

// ---------------------------------------------------------------------------
// Instance norm over L per (b,k)
// ---------------------------------------------------------------------------
__global__ __launch_bounds__(128)
void instnorm_kernel(const float* __restrict__ x, float* __restrict__ xn,
                     float* __restrict__ meanb, float* __restrict__ stdb)
{
  __shared__ float sbuf[128];
  const int bk = blockIdx.x;            // b*16 + k
  const int b = bk >> 4, k = bk & 15;
  const int tid = threadIdx.x;
  float v[4];
  #pragma unroll
  for (int i = 0; i < 4; ++i) {
    int t = tid + i * 128;
    v[i] = x[((size_t)b * L_CTX + t) * K_VAR + k];
  }
  float mu = block_sum(v[0] + v[1] + v[2] + v[3], sbuf) / (float)L_CTX;
  float sq = 0.f;
  #pragma unroll
  for (int i = 0; i < 4; ++i) { float d = v[i] - mu; sq += d * d; }
  float var = block_sum(sq, sbuf) / (float)L_CTX;
  float sd = sqrtf(var + EPSF);
  float inv = 1.f / sd;
  #pragma unroll
  for (int i = 0; i < 4; ++i) {
    int t = tid + i * 128;
    xn[(size_t)bk * L_CTX + t] = (v[i] - mu) * inv;
  }
  if (tid == 0) { meanb[bk] = mu; stdb[bk] = sd; }
}

// ---------------------------------------------------------------------------
// Patch embedding: token t = p*K_VAR + k
// ---------------------------------------------------------------------------
__global__ __launch_bounds__(128)
void patch_embed_kernel(const float* __restrict__ xn, const float* __restrict__ pw,
                        const float* __restrict__ pb, const float* __restrict__ pos,
                        const float* __restrict__ var, float* __restrict__ z)
{
  const int blk = blockIdx.x;                 // b*(K*P) + k*P + p
  const int b = blk / (K_VAR * P_PATCH);
  const int rem = blk % (K_VAR * P_PATCH);
  const int k = rem / P_PATCH;
  const int p = rem % P_PATCH;

  __shared__ float xv[16];
  if (threadIdx.x < 16)
    xv[threadIdx.x] = xn[((size_t)(b * K_VAR + k)) * L_CTX + p * 8 + threadIdx.x];
  __syncthreads();

  const size_t zrow = ((size_t)b * TOK_PER_B + p * K_VAR + k) * D_MODEL;
  for (int d = threadIdx.x; d < D_MODEL; d += 128) {
    float acc = pb[d] + pos[p * D_MODEL + d] + var[k * D_MODEL + d];
    #pragma unroll
    for (int j = 0; j < 16; ++j) acc += pw[d * 16 + j] * xv[j];
    z[zrow + d] = acc;
  }
}

// ---------------------------------------------------------------------------
// Selective scan with async-DMA B/C prefetch.
// Block: b = blockIdx.y, d = blockIdx.x*128 + tid. 16 states/lane in regs.
// Wave 0 prefetches row t+1's B|C (32 floats, 128B) into a 3-deep LDS ring
// via GLOBAL_LOAD_ASYNC_TO_LDS_B128 (lanes 0..7, 16B each), synced with
// s_wait_asynccnt; the DMA overlaps the step-t recurrence.
// ---------------------------------------------------------------------------
__global__ __launch_bounds__(128)
void scan_kernel(const float* __restrict__ u, const float* __restrict__ zg,
                 float* __restrict__ dly, const float* __restrict__ xdbl,
                 const float* __restrict__ A_log, const float* __restrict__ Dp)
{
  __shared__ __align__(16) float bcbuf[3][32];
  const int tid  = threadIdx.x;
  const int lane = tid & 31;
  const int wave = tid >> 5;
  const int b = blockIdx.y;
  const int d = blockIdx.x * 128 + tid;           // 0..1023
  const size_t base = (size_t)b * TOK_PER_B;

  auto issue_async = [&](int tt) {
    if (lane < 8) {
      const float* gsrc = &xdbl[(base + (size_t)tt) * 64 + DT_RANK + lane * 4];
      u32 ldst = (u32)(size_t)&bcbuf[tt % 3][lane * 4];   // LDS addr = addr[31:0]
      asm volatile("global_load_async_to_lds_b128 %0, %1, off"
                   :: "v"(ldst), "v"(gsrc) : "memory");
    }
  };

  float a[N_STATE], h[N_STATE];
  #pragma unroll
  for (int n = 0; n < N_STATE; ++n) {
    a[n] = -__expf(A_log[(size_t)d * N_STATE + n]);
    h[n] = 0.f;
  }
  const float dpv = Dp[d];

  if (wave == 0) issue_async(0);

  for (int t = 0; t < TOK_PER_B; ++t) {
    if (wave == 0) {
      if (t + 1 < TOK_PER_B) {
        issue_async(t + 1);                          // overlap with step t
        asm volatile("s_wait_asynccnt 0x1" ::: "memory");
      } else {
        asm volatile("s_wait_asynccnt 0x0" ::: "memory");
      }
    }
    __syncthreads();                                 // buf[t%3] visible to all

    const size_t row = base + t;
    const float dv = dly[row * D_INNER + d];
    const float uv = u[row * D_INNER + d];
    const float du = dv * uv;
    const float* bc = bcbuf[t % 3];                  // [B(16) | C(16)]
    float y = 0.f;
    #pragma unroll
    for (int n = 0; n < N_STATE; ++n) {
      float dA = __expf(dv * a[n]);
      h[n] = h[n] * dA + du * bc[n];
      y += h[n] * bc[N_STATE + n];
    }
    float zv  = zg[row * D_INNER + d];
    float sil = zv / (1.f + __expf(-zv));
    dly[row * D_INNER + d] = (y + uv * dpv) * sil;
  }
}

// ---------------------------------------------------------------------------
// Fused double layernorm: z = LN(t2; g1,b1), t2 = LN(z; g2,b2) (inplace)
// ---------------------------------------------------------------------------
__global__ __launch_bounds__(128)
void ln2_kernel(float* __restrict__ t2, float* __restrict__ z,
                const float* __restrict__ g1, const float* __restrict__ b1,
                const float* __restrict__ g2, const float* __restrict__ b2)
{
  __shared__ float sbuf[128];
  const size_t row = blockIdx.x;
  const int tid = threadIdx.x;
  float v[4];
  #pragma unroll
  for (int i = 0; i < 4; ++i) v[i] = t2[row * D_MODEL + tid + i * 128];

  float mu = block_sum(v[0] + v[1] + v[2] + v[3], sbuf) / (float)D_MODEL;
  float sq = 0.f;
  #pragma unroll
  for (int i = 0; i < 4; ++i) { float d = v[i] - mu; sq += d * d; }
  float r = rsqrtf(block_sum(sq, sbuf) / (float)D_MODEL + EPSF);

  float w[4];
  #pragma unroll
  for (int i = 0; i < 4; ++i) {
    int c = tid + i * 128;
    w[i] = (v[i] - mu) * r * g1[c] + b1[c];
    z[row * D_MODEL + c] = w[i];
  }
  float mu2 = block_sum(w[0] + w[1] + w[2] + w[3], sbuf) / (float)D_MODEL;
  float sq2 = 0.f;
  #pragma unroll
  for (int i = 0; i < 4; ++i) { float d = w[i] - mu2; sq2 += d * d; }
  float r2 = rsqrtf(block_sum(sq2, sbuf) / (float)D_MODEL + EPSF);
  #pragma unroll
  for (int i = 0; i < 4; ++i) {
    int c = tid + i * 128;
    t2[row * D_MODEL + c] = (w[i] - mu2) * r2 * g2[c] + b2[c];
  }
}

// ---------------------------------------------------------------------------
// Final LN fused with (B,P,K,D) -> (B,K,P*D) gather for the head GEMM
// ---------------------------------------------------------------------------
__global__ __launch_bounds__(128)
void final_ln_gather_kernel(const float* __restrict__ z, const float* __restrict__ g,
                            const float* __restrict__ bta, float* __restrict__ zhead)
{
  __shared__ float sbuf[128];
  const size_t row = blockIdx.x;                  // b*1008 + t
  const int b = (int)(row / TOK_PER_B);
  const int t = (int)(row % TOK_PER_B);
  const int p = t >> 4, k = t & 15;
  const int tid = threadIdx.x;
  float v[4];
  #pragma unroll
  for (int i = 0; i < 4; ++i) v[i] = z[row * D_MODEL + tid + i * 128];
  float mu = block_sum(v[0] + v[1] + v[2] + v[3], sbuf) / (float)D_MODEL;
  float sq = 0.f;
  #pragma unroll
  for (int i = 0; i < 4; ++i) { float d = v[i] - mu; sq += d * d; }
  float r = rsqrtf(block_sum(sq, sbuf) / (float)D_MODEL + EPSF);
  float* dst = zhead + ((size_t)(b * K_VAR + k)) * (P_PATCH * D_MODEL) + (size_t)p * D_MODEL;
  #pragma unroll
  for (int i = 0; i < 4; ++i) {
    int c = tid + i * 128;
    dst[c] = (v[i] - mu) * r * g[c] + bta[c];
  }
}

// ---------------------------------------------------------------------------
// Finalize: out[b, h, k] = pred[b*16+k, h] * std[bk] + mean[bk]
// ---------------------------------------------------------------------------
__global__ __launch_bounds__(256)
void finalize_kernel(const float* __restrict__ pred, const float* __restrict__ meanb,
                     const float* __restrict__ stdb, float* __restrict__ out)
{
  int i = blockIdx.x * 256 + threadIdx.x;
  if (i >= BATCH * H_PRED * K_VAR) return;
  int k  = i & 15;
  int hh = (i >> 4) % H_PRED;
  int b  = i / (H_PRED * K_VAR);
  int bk = b * K_VAR + k;
  out[i] = pred[(size_t)bk * H_PRED + hh] * stdb[bk] + meanb[bk];
}

// ---------------------------------------------------------------------------
static inline void launch_gemm(const float* A, const float* W, const float* bias,
                               const float* res, float* C,
                               int M, int N, int K, int lda, int act,
                               hipStream_t s)
{
  dim3 grid((N + TN - 1) / TN, (M + TM - 1) / TM);
  bool full = (M % TM == 0) && (N % TN == 0) && (K % TKS == 0);
  if (full)
    gemm_bf16_wmma<false><<<grid, 128, 0, s>>>(A, W, bias, res, C, M, N, K, lda, act);
  else
    gemm_bf16_wmma<true><<<grid, 128, 0, s>>>(A, W, bias, res, C, M, N, K, lda, act);
}

extern "C" void kernel_launch(void* const* d_in, const int* in_sizes, int n_in,
                              void* d_out, int out_size, void* d_ws, size_t ws_size,
                              hipStream_t stream)
{
  const float* x        = (const float*)d_in[0];
  const float* patch_w  = (const float*)d_in[1];
  const float* patch_b  = (const float*)d_in[2];
  const float* pos_emb  = (const float*)d_in[3];
  const float* var_emb  = (const float*)d_in[4];
  const float* in_proj  = (const float*)d_in[5];
  const float* x_proj   = (const float*)d_in[6];
  const float* dt_w     = (const float*)d_in[7];
  const float* dt_b     = (const float*)d_in[8];
  const float* A_log    = (const float*)d_in[9];
  const float* D_param  = (const float*)d_in[10];
  const float* out_proj = (const float*)d_in[11];
  const float* tmb_g    = (const float*)d_in[12];
  const float* tmb_b    = (const float*)d_in[13];
  const float* ffn_g    = (const float*)d_in[14];
  const float* ffn_b    = (const float*)d_in[15];
  const float* ffn_w1   = (const float*)d_in[16];
  const float* ffn_b1   = (const float*)d_in[17];
  const float* ffn_w2   = (const float*)d_in[18];
  const float* ffn_b2   = (const float*)d_in[19];
  const float* norm_g   = (const float*)d_in[20];
  const float* norm_b   = (const float*)d_in[21];
  const float* head_w   = (const float*)d_in[22];
  const float* head_b   = (const float*)d_in[23];

  float* ws = (float*)d_ws;
  const size_t T = T_TOT;
  float* xn    = ws;                                   // 131072
  float* meanb = xn + (size_t)BATCH * K_VAR * L_CTX;   // 256
  float* stdb  = meanb + 256;                          // 256
  float* z     = stdb + 256;                           // T*512
  float* u     = z + T * D_MODEL;                      // T*1024  (H region start)
  float* zg    = u + T * D_INNER;                      // T*1024
  float* dly   = zg + T * D_INNER;                     // T*1024  delta -> gated y
  float* xdbl  = dly + T * D_INNER;                    // T*64    (later: pred)
  float* t2    = xdbl + T * 64;                        // T*512
  float* H     = u;                                    // T*2048 spans u..zg
  float* zhead = u;                                    // 256*32256 (reuse)
  float* pred  = xdbl;                                 // 256*96 (reuse)

  instnorm_kernel<<<BATCH * K_VAR, 128, 0, stream>>>(x, xn, meanb, stdb);

  patch_embed_kernel<<<BATCH * K_VAR * P_PATCH, 128, 0, stream>>>(
      xn, patch_w, patch_b, pos_emb, var_emb, z);

  for (int l = 0; l < 3; ++l) {
    const float* Wxz  = in_proj  + (size_t)l * 2 * D_INNER * D_MODEL;
    const float* Wxd  = x_proj   + (size_t)l * (DT_RANK + 2 * N_STATE) * D_INNER;
    const float* Wdt  = dt_w     + (size_t)l * D_INNER * DT_RANK;
    const float* bdt  = dt_b     + (size_t)l * D_INNER;
    const float* Alog = A_log    + (size_t)l * D_INNER * N_STATE;
    const float* Dp   = D_param  + (size_t)l * D_INNER;
    const float* Wout = out_proj + (size_t)l * D_MODEL * D_INNER;
    const float* W1   = ffn_w1   + (size_t)l * D_FF * D_MODEL;
    const float* B1   = ffn_b1   + (size_t)l * D_FF;
    const float* W2   = ffn_w2   + (size_t)l * D_MODEL * D_FF;
    const float* B2   = ffn_b2   + (size_t)l * D_MODEL;

    launch_gemm(z, Wxz, nullptr, nullptr, u,
                T_TOT, D_INNER, D_MODEL, D_MODEL, ACT_SILU, stream);
    launch_gemm(z, Wxz + (size_t)D_INNER * D_MODEL, nullptr, nullptr, zg,
                T_TOT, D_INNER, D_MODEL, D_MODEL, ACT_NONE, stream);

    launch_gemm(u, Wxd, nullptr, nullptr, xdbl,
                T_TOT, DT_RANK + 2 * N_STATE, D_INNER, D_INNER, ACT_NONE, stream);

    launch_gemm(xdbl, Wdt, bdt, nullptr, dly,
                T_TOT, D_INNER, DT_RANK, 64, ACT_SOFTPLUS, stream);

    scan_kernel<<<dim3(D_INNER / 128, BATCH), 128, 0, stream>>>(
        u, zg, dly, xdbl, Alog, Dp);

    launch_gemm(dly, Wout, nullptr, z, t2,
                T_TOT, D_MODEL, D_INNER, D_INNER, ACT_NONE, stream);

    ln2_kernel<<<T_TOT, 128, 0, stream>>>(
        t2, z, tmb_g + l * D_MODEL, tmb_b + l * D_MODEL,
        ffn_g + l * D_MODEL, ffn_b + l * D_MODEL);

    launch_gemm(t2, W1, B1, nullptr, H,
                T_TOT, D_FF, D_MODEL, D_MODEL, ACT_GELU, stream);

    launch_gemm(H, W2, B2, z, z,
                T_TOT, D_MODEL, D_FF, D_FF, ACT_NONE, stream);
  }

  final_ln_gather_kernel<<<T_TOT, 128, 0, stream>>>(z, norm_g, norm_b, zhead);

  launch_gemm(zhead, head_w, head_b, nullptr, pred,
              BATCH * K_VAR, H_PRED, P_PATCH * D_MODEL, P_PATCH * D_MODEL,
              ACT_NONE, stream);

  finalize_kernel<<<(BATCH * H_PRED * K_VAR + 255) / 256, 256, 0, stream>>>(
      pred, meanb, stdb, (float*)d_out);
}
